// ScaledDotProductAttention_61624190763646
// MI455X (gfx1250) — matrix operand order repaired and checked by reading, not compiled, not run on hardware
//
#include <hip/hip_runtime.h>
#include <hip/hip_bf16.h>
#include <stdint.h>

// ---------------------------------------------------------------------------
// Scaled dot-product attention, B=16 S=4096 D=64, outputs (x, attn).
// attn write = 1 GiB -> HBM floor ~46us @ 23.3 TB/s. K/V (16 MB bf16) stay in L2,
// so QK^T recompute across passes is ~free. Three passes with *deferred*
// softmax reductions keep per-element VALU work minimal:
//   pass1: per-lane max only (1 v_max / element)
//   pass2: exp2 fma + per-lane sum + bf16 P + WMMA x accumulation
//   pass3: exp2 fma + non-temporal attn store (normalization folded into exp)
// ---------------------------------------------------------------------------

#define BB 16
#define SS 4096
#define DD 64
// exp(score/8) == exp2(score * K1), K1 = log2(e)/8
#define K1F 0.18033688011112042f

typedef __attribute__((ext_vector_type(16))) __bf16 v16bf;
typedef __attribute__((ext_vector_type(8)))  float  v8f;

union ABPack { v16bf bf; uint32_t u[8]; };

__device__ __forceinline__ uint32_t pack_bf16x2(float a, float b) {
  uint32_t ua = __float_as_uint(a), ub = __float_as_uint(b);
  ua = ua + 0x7FFFu + ((ua >> 16) & 1u);   // round-to-nearest-even
  ub = ub + 0x7FFFu + ((ub >> 16) & 1u);
  return (ua >> 16) | (ub & 0xFFFF0000u);
}

// ---- fp32 -> bf16 conversion of Q, K, V into workspace ----
__global__ void cvt_bf16_kernel(const float* __restrict__ Q,
                                const float* __restrict__ K,
                                const float* __restrict__ V,
                                uint32_t* __restrict__ Qb,
                                uint32_t* __restrict__ Kb,
                                uint32_t* __restrict__ Vb, int npair) {
  const float* src = (blockIdx.y == 0) ? Q : (blockIdx.y == 1) ? K : V;
  uint32_t*    dst = (blockIdx.y == 0) ? Qb : (blockIdx.y == 1) ? Kb : Vb;
  int i = blockIdx.x * blockDim.x + threadIdx.x;
  if (i < npair) {
    float2 f = ((const float2*)src)[i];
    dst[i] = pack_bf16x2(f.x, f.y);
  }
}

// K tile in LDS: 32 keys x 64 dims bf16 = 32 dwords/row, padded to 36
// (36 ≡ 0 mod 4 -> b128 copies; ln*36 mod 64 distinct -> conflict-free B read)
#define KSTR 36
// V^T / P patch rows padded to 17 dwords (gcd(17,64)=1 -> conflict-free)
#define TSTR 17

__device__ __forceinline__ v8f score_tile(const ABPack* aq, const uint32_t* sK,
                                          int s, int ln, int half) {
  v8f c = {0.f, 0.f, 0.f, 0.f, 0.f, 0.f, 0.f, 0.f};
#pragma unroll
  for (int cch = 0; cch < 2; ++cch) {       // D=64 -> two K=32 WMMA steps
    ABPack bm;                              // B = K^T : row=dim, col=key
#pragma unroll
    for (int v = 0; v < 8; ++v)
      bm.u[v] = sK[(s * 16 + ln) * KSTR + cch * 16 + half * 4 + ((v < 4) ? v : 8 + (v - 4))];
    c = __builtin_amdgcn_wmma_f32_16x16x32_bf16(
          false, aq[cch].bf, false, bm.bf, (short)0, c, false, false);
  }
  return c;
}

__device__ __forceinline__ void load_ktile(uint32_t* sK, const uint32_t* gk,
                                           int tid) {
#pragma unroll
  for (int r = 0; r < 2; ++r) {             // 4 KB tile as 256 x b128
    int i = tid + r * 128;
    uint4 g = ((const uint4*)gk)[i];
    *(uint4*)&sK[(i >> 3) * KSTR + (i & 7) * 4] = g;
  }
}

// ---- main attention kernel: 4 waves/WG, 16 queries per wave ----
__global__ __launch_bounds__(128)
void attn_kernel(const uint32_t* __restrict__ Qb,
                 const uint32_t* __restrict__ Kb,
                 const uint32_t* __restrict__ Vb,
                 float* __restrict__ Xout,
                 float* __restrict__ Attn) {
  __shared__ uint32_t sK[32 * KSTR];        // K tile
  __shared__ uint32_t sVt[64 * TSTR];       // V tile transposed [dim][key]
  __shared__ uint32_t sP[4][16 * TSTR];     // per-wave P patch [row][key] bf16

  const int tid  = threadIdx.x;
  const int wave = tid >> 5;
  const int lane = tid & 31;
  const int half = lane >> 4;               // half-wave (ISA A/B/C layouts)
  const int ln   = lane & 15;

  const int b  = blockIdx.x >> 6;           // 64 q-tiles per batch
  const int q0 = ((blockIdx.x & 63) << 6) + (wave << 4);

  // ---- Q as two A-matrices (16x32 bf16), ISA A layout:
  // lane(half,m): VGPR v holds K = (v<4 ? 2v : 16+2(v-4)) + 8*half (pairs)
  ABPack aq[2];
  {
    const uint32_t* qrow = Qb + (size_t)(b * SS + q0 + ln) * (DD / 2);
#pragma unroll
    for (int c = 0; c < 2; ++c)
#pragma unroll
      for (int v = 0; v < 8; ++v) {
        int kb2 = half * 4 + ((v < 4) ? v : 8 + (v - 4));
        aq[c].u[v] = qrow[c * 16 + kb2];
      }
  }

  const uint32_t* kBase = Kb + (size_t)b * SS * (DD / 2);
  const uint32_t* vBase = Vb + (size_t)b * SS * (DD / 2);

  // ================= PASS 1: per-lane row max (no shuffles, no exp) ========
  float mloc[8];
#pragma unroll
  for (int j = 0; j < 8; ++j) mloc[j] = -3.0e38f;

  for (int kt = 0; kt < SS; kt += 32) {
    load_ktile(sK, kBase + kt * 32, tid);
    if (kt + 32 < SS)   // overlap next tile fetch with compute
      __builtin_prefetch((const void*)(kBase + (kt + 32) * 32 + (tid << 3)), 0, 1);
    __syncthreads();
#pragma unroll
    for (int s = 0; s < 2; ++s) {
      v8f c = score_tile(aq, sK, s, ln, half);
#pragma unroll
      for (int j = 0; j < 8; ++j) mloc[j] = fmaxf(mloc[j], c[j]);
    }
    __syncthreads();
  }
  // one-time butterfly over the 16 lanes of each half-wave
  float m2[8];
#pragma unroll
  for (int j = 0; j < 8; ++j) {
    float mx = mloc[j];
    mx = fmaxf(mx, __shfl_xor(mx, 1, 16));
    mx = fmaxf(mx, __shfl_xor(mx, 2, 16));
    mx = fmaxf(mx, __shfl_xor(mx, 4, 16));
    mx = fmaxf(mx, __shfl_xor(mx, 8, 16));
    m2[j] = mx * K1F;                       // scale folded (scale > 0)
  }

  // ================= PASS 2: per-lane sum + unnormalized x = P_un*V ========
  float lloc[8];
#pragma unroll
  for (int j = 0; j < 8; ++j) lloc[j] = 0.0f;
  v8f o[4];
#pragma unroll
  for (int cc = 0; cc < 4; ++cc)
    o[cc] = (v8f){0.f, 0.f, 0.f, 0.f, 0.f, 0.f, 0.f, 0.f};

  for (int kt = 0; kt < SS; kt += 32) {
    const uint32_t* gv = vBase + kt * 32;
    load_ktile(sK, kBase + kt * 32, tid);
#pragma unroll
    for (int r = 0; r < 8; ++r) {           // V transposed into LDS [dim][key]
      int i = tid + r * 128;
      uint32_t d2 = gv[i];
      int key = i >> 5, dp = i & 31;
      uint16_t* vt = (uint16_t*)sVt;
      vt[(2 * dp) * (2 * TSTR) + key]     = (uint16_t)(d2 & 0xFFFFu);
      vt[(2 * dp + 1) * (2 * TSTR) + key] = (uint16_t)(d2 >> 16);
    }
    if (kt + 32 < SS) {
      __builtin_prefetch((const void*)(kBase + (kt + 32) * 32 + (tid << 3)), 0, 1);
      __builtin_prefetch((const void*)(vBase + (kt + 32) * 32 + (tid << 3)), 0, 1);
    }
    __syncthreads();

    __bf16* pw = (__bf16*)&sP[wave][0];
#pragma unroll
    for (int s = 0; s < 2; ++s) {
      v8f c = score_tile(aq, sK, s, ln, half);
#pragma unroll
      for (int j = 0; j < 8; ++j) {
        float p = __builtin_amdgcn_exp2f(c[j] * K1F - m2[j]);   // v_exp_f32
        lloc[j] += p;
        pw[(j + 8 * half) * (2 * TSTR) + s * 16 + ln] = (__bf16)p;
      }
    }
    __syncthreads();                        // P patch C-layout -> A-layout

    ABPack ap;
    {
      const uint32_t* pp = &sP[wave][0];
#pragma unroll
      for (int v = 0; v < 8; ++v) {
        int kb2 = half * 4 + ((v < 4) ? v : 8 + (v - 4));
        ap.u[v] = pp[ln * TSTR + kb2];
      }
    }
#pragma unroll
    for (int cc = 0; cc < 4; ++cc) {        // x[:, cc*16 .. cc*16+15]
      ABPack bv;                            // B = V : row=key, col=dim
#pragma unroll
      for (int v = 0; v < 8; ++v)
        bv.u[v] = sVt[(cc * 16 + ln) * TSTR + half * 4 + ((v < 4) ? v : 8 + (v - 4))];
      o[cc] = __builtin_amdgcn_wmma_f32_16x16x32_bf16(
                false, ap.bf, false, bv.bf, (short)0, o[cc], false, false);
    }
    __syncthreads();
  }

  // one-time butterfly row-sum; fold normalization into exponent for pass 3
  float m3[8], linv[8];
#pragma unroll
  for (int j = 0; j < 8; ++j) {
    float su = lloc[j];
    su += __shfl_xor(su, 1, 16);
    su += __shfl_xor(su, 2, 16);
    su += __shfl_xor(su, 4, 16);
    su += __shfl_xor(su, 8, 16);
    m3[j]   = m2[j] + __builtin_amdgcn_logf(su);   // v_log_f32 (base 2)
    linv[j] = 1.0f / su;
  }

  // ---- store x = O / l ----
  {
    float* xB = Xout + (size_t)(b * SS + q0) * DD;
#pragma unroll
    for (int cc = 0; cc < 4; ++cc)
#pragma unroll
      for (int j = 0; j < 8; ++j)
        __builtin_nontemporal_store(
            o[cc][j] * linv[j],
            &xB[(j + 8 * half) * DD + cc * 16 + ln]);
  }

  // ================= PASS 3: stream normalized attn (1 GiB, NT) ============
  float* attnB = Attn + (size_t)b * SS * SS + (size_t)q0 * SS;
  for (int kt = 0; kt < SS; kt += 32) {
    load_ktile(sK, kBase + kt * 32, tid);
    if (kt + 32 < SS)
      __builtin_prefetch((const void*)(kBase + (kt + 32) * 32 + (tid << 3)), 0, 1);
    __syncthreads();
#pragma unroll
    for (int s = 0; s < 2; ++s) {
      v8f c = score_tile(aq, sK, s, ln, half);
#pragma unroll
      for (int j = 0; j < 8; ++j) {
        float p = __builtin_amdgcn_exp2f(c[j] * K1F - m3[j]);  // already /l
        __builtin_nontemporal_store(
            p, &attnB[(size_t)(j + 8 * half) * SS + kt + s * 16 + ln]);
      }
    }
    __syncthreads();
  }
}

extern "C" void kernel_launch(void* const* d_in, const int* in_sizes, int n_in,
                              void* d_out, int out_size, void* d_ws, size_t ws_size,
                              hipStream_t stream) {
  (void)in_sizes; (void)n_in; (void)out_size; (void)ws_size;
  const float* Q = (const float*)d_in[0];
  const float* K = (const float*)d_in[1];
  const float* V = (const float*)d_in[2];

  float* X    = (float*)d_out;                          // [B,S,D]
  float* Attn = (float*)d_out + (size_t)BB * SS * DD;   // [B,S,S]

  const int n = BB * SS * DD;          // elements per tensor (4.19M)
  uint32_t* ws = (uint32_t*)d_ws;      // needs 3 * n * 2 bytes = 24 MB
  uint32_t* Qb = ws;
  uint32_t* Kb = ws + (size_t)n / 2;
  uint32_t* Vb = ws + (size_t)n;

  const int npair = n / 2;
  dim3 cgrid((npair + 255) / 256, 3);
  cvt_bf16_kernel<<<cgrid, 256, 0, stream>>>(Q, K, V, Qb, Kb, Vb, npair);

  attn_kernel<<<dim3(BB * (SS / 64)), dim3(128), 0, stream>>>(Qb, Kb, Vb, X, Attn);
}
